// SemiSupervisedGAT_43499428774652
// MI455X (gfx1250) — matrix operand order repaired and checked by reading, not compiled, not run on hardware
//
#include <hip/hip_runtime.h>
#include <hip/hip_bf16.h>

#define NNODES   100000
#define NEDGES   1600000
#define IN_DIMS  128
#define HIDD     64
#define HEAD_DIM 32

typedef __attribute__((ext_vector_type(16))) __bf16 v16bf;
typedef __attribute__((ext_vector_type(8)))  float  v8f;

// ---------- helpers -------------------------------------------------------

__device__ __forceinline__ float waveReduceMax(float v) {
  #pragma unroll
  for (int off = 16; off > 0; off >>= 1) v = fmaxf(v, __shfl_xor(v, off, 32));
  return v;
}
__device__ __forceinline__ float waveReduceSum(float v) {
  #pragma unroll
  for (int off = 16; off > 0; off >>= 1) v += __shfl_xor(v, off, 32);
  return v;
}

// order-preserving encoding of f32 into u32 so atomicMax works for any sign
__device__ __forceinline__ unsigned int encodeOrderedF32(float f) {
  unsigned int u = __float_as_uint(f);
  return (u & 0x80000000u) ? ~u : (u | 0x80000000u);
}
__device__ __forceinline__ float decodeOrderedF32(unsigned int e) {
  unsigned int u = (e & 0x80000000u) ? (e ^ 0x80000000u) : ~e;
  return __uint_as_float(u);
}
// encodeOrderedF32(-inf) == 0x007FFFFF
#define ENC_NEG_INF 0x007FFFFFu

// ---------- init ----------------------------------------------------------

__global__ void init_kernel(float* __restrict__ acc, unsigned int* __restrict__ gmax,
                            float* __restrict__ gsum, int total) {
  const int i = blockIdx.x * 256 + threadIdx.x;
  if (i < total) acc[i] = 0.0f;
  if (i < 4) { gmax[i] = ENC_NEG_INF; gsum[i] = 0.0f; }
}

// ---------- WMMA GEMM: H[N,64] = X[N,K] @ W[K,64] + bias ------------------
// 256 threads = 8 wave32; wave w -> Mtile (w>>2), Ntile (w&3). Block = 32 rows.

template <int K>
__global__ __launch_bounds__(256) void gemm_bias_wmma(
    const float* __restrict__ X, const float* __restrict__ W,
    const float* __restrict__ bias, float* __restrict__ H, int nrows)
{
  __shared__ float sW[K * HIDD];
  const int tid = threadIdx.x;
  for (int i = tid; i < K * HIDD; i += 256) sW[i] = W[i];
  __syncthreads();

  const int lane = tid & 31;
  const int wave = tid >> 5;
  const int row0 = blockIdx.x * 32 + (wave >> 2) * 16;
  if (row0 >= nrows) return;
  const int nt = (wave & 3) * 16;

  const int mA  = row0 + (lane & 15);       // A row for this lane
  const int kAh = (lane >> 4) * 8;          // A K-half select (ISA 16-bit A layout)
  const int nB  = nt + (lane & 15);         // B column for this lane
  const int kBh = (lane >> 4) * 8;          // B K-half select

  v8f c = {};
  #pragma unroll
  for (int kk = 0; kk < K; kk += 32) {
    v16bf a, b;
    #pragma unroll
    for (int v = 0; v < 8; ++v) {
      // A 16x32 bf16: VGPR v holds K = 2*(v&3) + 16*(v>>2) + (lane>=16 ? 8 : 0), pair lo/hi
      const int ka = kk + ((v & 3) * 2) + ((v >> 2) * 16) + kAh;
      a[2 * v]     = (__bf16)X[(size_t)mA * K + ka];
      a[2 * v + 1] = (__bf16)X[(size_t)mA * K + ka + 1];
      // B 32x16 bf16: VGPR v holds K = 2*(v&3) + 16*(v>>2) + (lane>=16 ? 8 : 0), pair lo/hi, N = lane&15
      const int kb = kk + ((v & 3) * 2) + ((v >> 2) * 16) + kBh;
      b[2 * v]     = (__bf16)sW[kb * HIDD + nB];
      b[2 * v + 1] = (__bf16)sW[(kb + 1) * HIDD + nB];
    }
    c = __builtin_amdgcn_wmma_f32_16x16x32_bf16(false, a, false, b, (short)0, c,
                                                false, false);
  }

  // D 16x16 f32: VGPR v -> row = v + (lane>=16 ? 8 : 0), col = lane&15
  const int   mD = row0 + ((lane >> 4) * 8);
  const float bn = bias[nB];
  #pragma unroll
  for (int v = 0; v < 8; ++v)
    if (mD + v < nrows) H[(size_t)(mD + v) * HIDD + nB] = c[v] + bn;
}

// ---------- edge attention logits + global per-head max -------------------

__global__ __launch_bounds__(256) void edge_attn_kernel(
    const float* __restrict__ h, const int* __restrict__ src, const int* __restrict__ dst,
    const float* __restrict__ A, const float* __restrict__ ab,
    float* __restrict__ attn, unsigned int* __restrict__ gmax, int nedges)
{
  __shared__ float sA[2 * HIDD * 2];  // [128,2] row-major
  __shared__ float red0[8], red1[8];
  for (int i = threadIdx.x; i < 2 * HIDD * 2; i += 256) sA[i] = A[i];
  __syncthreads();

  const int e = blockIdx.x * 256 + threadIdx.x;
  float a0 = -__builtin_inff(), a1 = -__builtin_inff();
  if (e < nedges) {
    const int s = src[e], d = dst[e];
    const float* hs = h + (size_t)s * HIDD;
    const float* hd = h + (size_t)d * HIDD;
    float acc0 = ab[0], acc1 = ab[1];
    #pragma unroll 4
    for (int i = 0; i < HIDD; ++i) {
      const float vs = hs[i], vd = hd[i];
      acc0 += vs * sA[i * 2 + 0] + vd * sA[(HIDD + i) * 2 + 0];
      acc1 += vs * sA[i * 2 + 1] + vd * sA[(HIDD + i) * 2 + 1];
    }
    attn[(size_t)e * 2 + 0] = acc0;
    attn[(size_t)e * 2 + 1] = acc1;
    a0 = acc0; a1 = acc1;
  }
  a0 = waveReduceMax(a0);
  a1 = waveReduceMax(a1);
  if ((threadIdx.x & 31) == 0) { red0[threadIdx.x >> 5] = a0; red1[threadIdx.x >> 5] = a1; }
  __syncthreads();
  if (threadIdx.x == 0) {
    float m0 = red0[0], m1 = red1[0];
    #pragma unroll
    for (int w = 1; w < 8; ++w) { m0 = fmaxf(m0, red0[w]); m1 = fmaxf(m1, red1[w]); }
    atomicMax(gmax + 0, encodeOrderedF32(m0));
    atomicMax(gmax + 1, encodeOrderedF32(m1));
  }
}

// ---------- exp(attn - max) in place + global per-head sum ----------------

__global__ __launch_bounds__(256) void edge_exp_kernel(
    float* __restrict__ attn, const unsigned int* __restrict__ gmax,
    float* __restrict__ gsum, int nedges)
{
  __shared__ float red0[8], red1[8];
  const float m0 = decodeOrderedF32(gmax[0]);
  const float m1 = decodeOrderedF32(gmax[1]);
  const int e = blockIdx.x * 256 + threadIdx.x;
  float s0 = 0.0f, s1 = 0.0f;
  if (e < nedges) {
    s0 = __expf(attn[(size_t)e * 2 + 0] - m0);
    s1 = __expf(attn[(size_t)e * 2 + 1] - m1);
    attn[(size_t)e * 2 + 0] = s0;
    attn[(size_t)e * 2 + 1] = s1;
  }
  s0 = waveReduceSum(s0);
  s1 = waveReduceSum(s1);
  if ((threadIdx.x & 31) == 0) { red0[threadIdx.x >> 5] = s0; red1[threadIdx.x >> 5] = s1; }
  __syncthreads();
  if (threadIdx.x == 0) {
    float t0 = 0.0f, t1 = 0.0f;
    #pragma unroll
    for (int w = 0; w < 8; ++w) { t0 += red0[w]; t1 += red1[w]; }
    atomicAdd(gsum + 0, t0);
    atomicAdd(gsum + 1, t1);
  }
}

// ---------- scatter: acc[dst] += softmax_w * src_h[:32] (both heads) ------

__global__ __launch_bounds__(256) void edge_scatter_kernel(
    const float* __restrict__ h, const int* __restrict__ src, const int* __restrict__ dst,
    const float* __restrict__ attn, const float* __restrict__ gsum,
    float* __restrict__ acc, int nedges)
{
  const int e = blockIdx.x * 256 + threadIdx.x;
  if (e >= nedges) return;
  const float w0 = attn[(size_t)e * 2 + 0] / gsum[0];
  const float w1 = attn[(size_t)e * 2 + 1] / gsum[1];
  const int s = src[e], d = dst[e];
  const float* hs = h + (size_t)s * HIDD;   // reference uses first HEAD_DIM cols for BOTH heads
  float* ad = acc + (size_t)d * HIDD;
  #pragma unroll 8
  for (int i = 0; i < HEAD_DIM; ++i) {
    const float v = hs[i];
    atomicAdd(ad + i, w0 * v);
    atomicAdd(ad + HEAD_DIM + i, w1 * v);
  }
}

// ---------- out = relu(acc + h); acc re-zeroed for next layer -------------

__global__ void combine_relu_kernel(float* __restrict__ acc, const float* __restrict__ h,
                                    float* __restrict__ out, int total)
{
  const int i = blockIdx.x * 256 + threadIdx.x;
  if (i < total) {
    const float v = acc[i] + h[i];
    out[i] = v > 0.0f ? v : 0.0f;
    acc[i] = 0.0f;
  }
}

// ---------- classifier: relu(x@Wc1+bc1)@Wc2+bc2 ---------------------------

__global__ __launch_bounds__(256) void classifier_kernel(
    const float* __restrict__ x, const float* __restrict__ Wc1, const float* __restrict__ bc1,
    const float* __restrict__ Wc2, const float* __restrict__ bc2,
    float* __restrict__ out, int n)
{
  __shared__ float sW1[HIDD * 32];
  __shared__ float sW2[32 * 2];
  __shared__ float sb1[32];
  __shared__ float sb2[2];
  for (int i = threadIdx.x; i < HIDD * 32; i += 256) sW1[i] = Wc1[i];
  if (threadIdx.x < 64) sW2[threadIdx.x] = Wc2[threadIdx.x];
  if (threadIdx.x < 32) sb1[threadIdx.x] = bc1[threadIdx.x];
  if (threadIdx.x < 2)  sb2[threadIdx.x] = bc2[threadIdx.x];
  __syncthreads();

  const int node = blockIdx.x * 256 + threadIdx.x;
  if (node >= n) return;
  const float* xr = x + (size_t)node * HIDD;
  float xv[HIDD];
  #pragma unroll
  for (int i = 0; i < HIDD; ++i) xv[i] = xr[i];

  float l0 = sb2[0], l1 = sb2[1];
  #pragma unroll 4
  for (int j = 0; j < 32; ++j) {
    float hj = sb1[j];
    #pragma unroll 8
    for (int i = 0; i < HIDD; ++i) hj += xv[i] * sW1[i * 32 + j];
    hj = hj > 0.0f ? hj : 0.0f;
    l0 += hj * sW2[j * 2 + 0];
    l1 += hj * sW2[j * 2 + 1];
  }
  out[(size_t)node * 2 + 0] = l0;
  out[(size_t)node * 2 + 1] = l1;
}

// ---------- launch --------------------------------------------------------

extern "C" void kernel_launch(void* const* d_in, const int* in_sizes, int n_in,
                              void* d_out, int out_size, void* d_ws, size_t ws_size,
                              hipStream_t stream) {
  const float* features = (const float*)d_in[0];
  const int*   edge     = (const int*)d_in[1];   // [2, E] row-major
  const float* W0  = (const float*)d_in[2];
  const float* b0  = (const float*)d_in[3];
  const float* A0  = (const float*)d_in[4];
  const float* a0  = (const float*)d_in[5];
  const float* W1  = (const float*)d_in[6];
  const float* b1  = (const float*)d_in[7];
  const float* A1  = (const float*)d_in[8];
  const float* a1  = (const float*)d_in[9];
  const float* Wc1 = (const float*)d_in[10];
  const float* bc1 = (const float*)d_in[11];
  const float* Wc2 = (const float*)d_in[12];
  const float* bc2 = (const float*)d_in[13];
  const int* src = edge;
  const int* dst = edge + NEDGES;

  const size_t N64 = (size_t)NNODES * HIDD;
  float* ws    = (float*)d_ws;
  float* slotA = ws;                 // h0 / h1            (N*64)
  float* acc   = ws + N64;           // scatter accumulator (N*64)
  float* slotC = ws + 2 * N64;       // x1 / x2             (N*64)
  float* attn  = ws + 3 * N64;       // edge attn           (E*2)
  unsigned int* gmax = (unsigned int*)(ws + 3 * N64 + 2 * (size_t)NEDGES); // 4
  float* gsum = (float*)(gmax + 4);                                        // 4

  const int egrid   = (NEDGES + 255) / 256;        // 6250
  const int ngrid64 = (int)((N64 + 255) / 256);
  const int mgrid   = NNODES / 32;                 // 3125 (exact)
  const int ngrid   = (NNODES + 255) / 256;

  init_kernel<<<ngrid64, 256, 0, stream>>>(acc, gmax, gsum, (int)N64);

  // ---- layer 0 ----
  gemm_bias_wmma<IN_DIMS><<<mgrid, 256, 0, stream>>>(features, W0, b0, slotA, NNODES);
  edge_attn_kernel<<<egrid, 256, 0, stream>>>(slotA, src, dst, A0, a0, attn, gmax + 0, NEDGES);
  edge_exp_kernel<<<egrid, 256, 0, stream>>>(attn, gmax + 0, gsum + 0, NEDGES);
  edge_scatter_kernel<<<egrid, 256, 0, stream>>>(slotA, src, dst, attn, gsum + 0, acc, NEDGES);
  combine_relu_kernel<<<ngrid64, 256, 0, stream>>>(acc, slotA, slotC, (int)N64);

  // ---- layer 1 ----
  gemm_bias_wmma<HIDD><<<mgrid, 256, 0, stream>>>(slotC, W1, b1, slotA, NNODES);
  edge_attn_kernel<<<egrid, 256, 0, stream>>>(slotA, src, dst, A1, a1, attn, gmax + 2, NEDGES);
  edge_exp_kernel<<<egrid, 256, 0, stream>>>(attn, gmax + 2, gsum + 2, NEDGES);
  edge_scatter_kernel<<<egrid, 256, 0, stream>>>(slotA, src, dst, attn, gsum + 2, acc, NEDGES);
  combine_relu_kernel<<<ngrid64, 256, 0, stream>>>(acc, slotA, slotC, (int)N64);

  // ---- classifier ----
  classifier_kernel<<<ngrid, 256, 0, stream>>>(slotC, Wc1, bc1, Wc2, bc2, (float*)d_out, NNODES);
}
